// StatefulMambaBlock_5540507812262
// MI455X (gfx1250) — matrix operand order repaired and checked by reading, not compiled, not run on hardware
//
#include <hip/hip_runtime.h>
#include <hip/hip_bf16.h>

typedef __attribute__((ext_vector_type(16))) _Float16 v16h;
typedef __attribute__((ext_vector_type(8)))  _Float16 v8h;
typedef __attribute__((ext_vector_type(8)))  float    v8f;

// Problem constants
static constexpr int B_   = 2;
static constexpr int L_   = 1024;
static constexpr int D_   = 1024;
static constexpr int DI_  = 2048;   // D_INNER
static constexpr int E2_  = 4096;   // 2*D_INNER
static constexpr int NTOK = B_ * L_;        // 2048 tokens
static constexpr int DTR  = 64;     // DT_RANK
static constexpr int NS   = 16;     // N_STATE

// LDS B-panel geometry for the staged GEMM
static constexpr int NBLK    = 64;   // N columns per block
static constexpr int MBLK    = 128;  // M rows per block (8 waves x 16)
static constexpr int BSTRIDE = 40;   // halves per LDS row (32 used + 8 pad = 80B, bank-spread)

// ---------------------------------------------------------------------------
// f32 -> f16 conversion (weights + activations)
// ---------------------------------------------------------------------------
__global__ __launch_bounds__(256) void f32_to_f16_kernel(const float* __restrict__ src,
                                                         _Float16* __restrict__ dst, int n) {
  int i = blockIdx.x * blockDim.x + threadIdx.x;
  if (i < n) dst[i] = (_Float16)src[i];
}

// ---------------------------------------------------------------------------
// LayerNorm over D=1024, one block per token, outputs f16
// ---------------------------------------------------------------------------
__global__ __launch_bounds__(256) void ln_kernel(const float* __restrict__ x,
                                                 const float* __restrict__ w,
                                                 const float* __restrict__ bia,
                                                 _Float16* __restrict__ xn) {
  const int tok = blockIdx.x;
  const float* xr = x + (size_t)tok * D_;
  float v[4];
  float s = 0.f, s2 = 0.f;
#pragma unroll
  for (int i = 0; i < 4; ++i) {
    v[i] = xr[threadIdx.x + 256 * i];
    s += v[i];
    s2 += v[i] * v[i];
  }
#pragma unroll
  for (int off = 16; off; off >>= 1) {
    s  += __shfl_xor(s, off, 32);
    s2 += __shfl_xor(s2, off, 32);
  }
  __shared__ float ss[8], ss2[8];
  __shared__ float mu_s, rinv_s;
  const int wid = threadIdx.x >> 5;
  if ((threadIdx.x & 31) == 0) { ss[wid] = s; ss2[wid] = s2; }
  __syncthreads();
  if (threadIdx.x == 0) {
    float t = 0.f, t2 = 0.f;
#pragma unroll
    for (int i = 0; i < 8; ++i) { t += ss[i]; t2 += ss2[i]; }
    const float mu  = t  * (1.f / D_);
    const float var = t2 * (1.f / D_) - mu * mu;
    mu_s   = mu;
    rinv_s = rsqrtf(var + 1e-5f);
  }
  __syncthreads();
  const float mu = mu_s, rinv = rinv_s;
#pragma unroll
  for (int i = 0; i < 4; ++i) {
    const int c = threadIdx.x + 256 * i;
    xn[(size_t)tok * D_ + c] = (_Float16)((v[i] - mu) * rinv * w[c] + bia[c]);
  }
}

// ---------------------------------------------------------------------------
// Fragment helpers (wave32 WMMA layouts)
// A fragment (16x32 f16): lane<16 -> M=lane, K in {0..7, 16..23};
//                         lane>=16 -> M=lane-16, K in {8..15, 24..31}.
// B fragment (32x16 f16): lane<16 -> N=lane, K in {0..7, 16..23};
//                         lane>=16 -> N=lane-16, K in {8..15, 24..31}.
// D/C (16x16 f32, 8 VGPRs): VGPR j, lane<16 -> M=j, N=lane;
//                                    lane>=16 -> M=j+8, N=lane-16.
// ---------------------------------------------------------------------------
__device__ __forceinline__ v16h cat16(v8h lo, v8h hi) {
  return __builtin_shufflevector(lo, hi, 0, 1, 2, 3, 4, 5, 6, 7,
                                         8, 9, 10, 11, 12, 13, 14, 15);
}

// ---------------------------------------------------------------------------
// LDS-staged, double-buffered WMMA GEMM:
//   C[M,N] = A[M,K] * Bw[N,K]^T (+ optional residual), f16 in / f32 out.
// Block = 256 threads = 8 waves; macro-tile 128(M) x 64(N).
// Wave w computes rows [m0 + 16w, m0 + 16w + 16), all waves share the
// 64-row B panel staged in LDS (double buffer, 1 barrier per k-step).
// All 8 ds_load_b128 for the 4 B fragments are issued before the 4 WMMAs
// so the dscnt waits retire partially and LDS latency overlaps the math.
// ---------------------------------------------------------------------------
template <bool ADD_RES>
__global__ __launch_bounds__(256) void wmma_gemm_lds_kernel(const _Float16* __restrict__ A,
                                                            const _Float16* __restrict__ Bw,
                                                            float* __restrict__ C,
                                                            const float* __restrict__ Res,
                                                            int M, int N, int K) {
  __shared__ _Float16 Bs[2][NBLK * BSTRIDE];

  const int nbCount = N / NBLK;
  const int mb = blockIdx.x / nbCount;
  const int nb = blockIdx.x - mb * nbCount;
  const int wave = threadIdx.x >> 5;
  const int lane = threadIdx.x & 31;
  const int sub = lane & 15;
  const bool hi = lane >= 16;

  const int m0 = mb * MBLK + wave * 16;   // this wave's M rows
  const int n0 = nb * NBLK;               // block's N columns

  // Staging assignment: thread t loads 16B of the B panel per k-step.
  // row = t>>2 (0..63), chunk = t&3 (8 halves each).
  const int srow = threadIdx.x >> 2;
  const int schunk = (threadIdx.x & 3) * 8;
  const _Float16* sGlob = Bw + (size_t)(n0 + srow) * K + schunk;
  _Float16* sLds0 = &Bs[0][srow * BSTRIDE + schunk];
  _Float16* sLds1 = &Bs[1][srow * BSTRIDE + schunk];

  // Stage k-step 0 into buffer 0
  *(v8h*)sLds0 = *(const v8h*)sGlob;
  __syncthreads();

  v8f acc[4] = {};
  const _Float16* aRow = A + (size_t)(m0 + sub) * K + (hi ? 8 : 0);
  const int bko = hi ? 8 : 0;

  const int ksteps = K >> 5;
  for (int ks = 0; ks < ksteps; ++ks) {
    const int k0 = ks << 5;
    const int cur = ks & 1;

    // Stage next k-step into the alternate buffer (overlaps the WMMAs below)
    if (ks + 1 < ksteps) {
      v8h nxt = *(const v8h*)(sGlob + k0 + 32);
      *(v8h*)(cur ? sLds0 : sLds1) = nxt;
      __builtin_prefetch(sGlob + k0 + 64, 0, 1);
    }

    // A fragment from global
    v8h alo = *(const v8h*)(aRow + k0);
    v8h ahi = *(const v8h*)(aRow + k0 + 16);
    if (ks + 1 < ksteps) __builtin_prefetch(aRow + k0 + 32, 0, 1);
    v16h a = cat16(alo, ahi);

    // Issue all 8 LDS fragment loads before any WMMA
    v16h b[4];
#pragma unroll
    for (int i = 0; i < 4; ++i) {
      const _Float16* bp = &Bs[cur][(i * 16 + sub) * BSTRIDE + bko];
      b[i] = cat16(*(const v8h*)bp, *(const v8h*)(bp + 16));
    }
#pragma unroll
    for (int i = 0; i < 4; ++i)
      acc[i] = __builtin_amdgcn_wmma_f32_16x16x32_f16(
          false, a, false, b[i], (short)0, acc[i], false, false);

    __syncthreads();
  }

  // Store D (+ optional residual)
  const int rbase = m0 + (hi ? 8 : 0);
#pragma unroll
  for (int i = 0; i < 4; ++i) {
#pragma unroll
    for (int j = 0; j < 8; ++j) {
      const size_t idx = (size_t)(rbase + j) * N + n0 + i * 16 + sub;
      float vv = acc[i][j];
      if (ADD_RES) vv += Res[idx];
      C[idx] = vv;
    }
  }
}

// ---------------------------------------------------------------------------
// Direct (no-LDS) WMMA GEMM for the small N=96 projection.
// Each wave computes a 16 x (16*NW) strip.
// ---------------------------------------------------------------------------
template <int NW>
__global__ __launch_bounds__(256) void wmma_gemm_direct_kernel(const _Float16* __restrict__ A,
                                                               const _Float16* __restrict__ Bw,
                                                               float* __restrict__ C,
                                                               int M, int N, int K) {
  const int wave = blockIdx.x * (blockDim.x >> 5) + (threadIdx.x >> 5);
  const int lane = threadIdx.x & 31;
  const int ntiles = N / (16 * NW);
  const int mt = wave / ntiles;
  const int nt = wave % ntiles;
  if (mt >= M / 16) return;

  const int m0 = mt * 16;
  const int n0 = nt * 16 * NW;
  const int sub = lane & 15;
  const bool hi = lane >= 16;

  v8f acc[NW] = {};
  const _Float16* aRow = A + (size_t)(m0 + sub) * K + (hi ? 8 : 0);
  const int bko = hi ? 8 : 0;

  for (int k0 = 0; k0 < K; k0 += 32) {
    if (k0 + 32 < K) __builtin_prefetch(aRow + k0 + 32, 0, 1);
    v16h a = cat16(*(const v8h*)(aRow + k0), *(const v8h*)(aRow + k0 + 16));
    // Load all B fragments before the WMMAs so waits can retire partially.
    v16h b[NW];
#pragma unroll
    for (int i = 0; i < NW; ++i) {
      const _Float16* bp = Bw + (size_t)(n0 + i * 16 + sub) * K + k0 + bko;
      b[i] = cat16(*(const v8h*)bp, *(const v8h*)(bp + 16));
    }
#pragma unroll
    for (int i = 0; i < NW; ++i)
      acc[i] = __builtin_amdgcn_wmma_f32_16x16x32_f16(
          false, a, false, b[i], (short)0, acc[i], false, false);
  }

  const int rbase = m0 + (hi ? 8 : 0);
#pragma unroll
  for (int i = 0; i < NW; ++i) {
#pragma unroll
    for (int j = 0; j < 8; ++j) {
      C[(size_t)(rbase + j) * N + n0 + i * 16 + sub] = acc[i][j];
    }
  }
}

// ---------------------------------------------------------------------------
// Causal depthwise conv (K=4) along L + SiLU; outputs f32 and f16 copies
// ---------------------------------------------------------------------------
__global__ __launch_bounds__(256) void conv_silu_kernel(const float* __restrict__ xz,
                                                        const float* __restrict__ Wc,
                                                        const float* __restrict__ bc,
                                                        float* __restrict__ xc,
                                                        _Float16* __restrict__ xch) {
  const int idx = blockIdx.x * blockDim.x + threadIdx.x;  // B*L*DI
  if (idx >= B_ * L_ * DI_) return;
  const int e = idx & (DI_ - 1);
  const int l = (idx >> 11) & (L_ - 1);
  const int b = idx >> 21;
  float acc = bc[e];
#pragma unroll
  for (int k = 0; k < 4; ++k) {
    const int ls = l + k - 3;
    if (ls >= 0)
      acc += xz[((size_t)(b * L_ + ls)) * E2_ + e] * Wc[e * 4 + k];
  }
  const float sv = acc / (1.f + __expf(-acc));
  xc[idx] = sv;
  xch[idx] = (_Float16)sv;
}

// ---------------------------------------------------------------------------
// Split dbl (NTOK x 96) into dt_low(f16, 64), B_ssm(f32, 16), C_ssm(f32, 16)
// ---------------------------------------------------------------------------
__global__ __launch_bounds__(256) void pack_kernel(const float* __restrict__ dbl,
                                                   _Float16* __restrict__ dtl,
                                                   float* __restrict__ Bs,
                                                   float* __restrict__ Cs) {
  const int idx = blockIdx.x * blockDim.x + threadIdx.x;  // NTOK*96
  if (idx >= NTOK * 96) return;
  const int m = idx / 96;
  const int r = idx - m * 96;
  const float v = dbl[idx];
  if (r < DTR)            dtl[m * DTR + r] = (_Float16)v;
  else if (r < DTR + NS)  Bs[m * NS + (r - DTR)] = v;
  else                    Cs[m * NS + (r - DTR - NS)] = v;
}

// ---------------------------------------------------------------------------
// Selective scan: 16 lanes per (b,e) channel, lane n owns state n.
// y_t = sum_n h_n C_n via __shfl_xor intra-16-lane reduce.
// Output gated y in f16 for the final WMMA GEMM.
// ---------------------------------------------------------------------------
__global__ __launch_bounds__(256) void scan_kernel(const float* __restrict__ dpre,
                                                   const float* __restrict__ xc,
                                                   const float* __restrict__ Bs,
                                                   const float* __restrict__ Cs,
                                                   const float* __restrict__ xz,
                                                   const float* __restrict__ Alog,
                                                   const float* __restrict__ bdt,
                                                   const float* __restrict__ Dskip,
                                                   _Float16* __restrict__ yh) {
  const int t = blockIdx.x * blockDim.x + threadIdx.x;  // B*DI*NS = 65536
  const int n  = t & (NS - 1);
  const int ch = t >> 4;           // 0..4095
  const int e  = ch & (DI_ - 1);
  const int b  = ch >> 11;

  const float An = -__expf(Alog[e * NS + n]);
  const float bd = bdt[e];
  const float Dv = Dskip[e];
  float h = 0.f;

  for (int l = 0; l < L_; ++l) {
    const size_t base = (size_t)(b * L_ + l);
    const float dp = dpre[base * DI_ + e] + bd;
    const float delta = (dp > 20.f) ? dp : log1pf(__expf(dp));
    const float xv = xc[base * DI_ + e];
    const float Bn = Bs[base * NS + n];
    const float Cn = Cs[base * NS + n];
    h = __expf(delta * An) * h + delta * Bn * xv;
    float py = h * Cn;
    py += __shfl_xor(py, 1, 16);
    py += __shfl_xor(py, 2, 16);
    py += __shfl_xor(py, 4, 16);
    py += __shfl_xor(py, 8, 16);
    if (n == 0) {
      const float zv = xz[base * E2_ + DI_ + e];
      const float y = (py + Dv * xv) * (zv / (1.f + __expf(-zv)));
      yh[base * DI_ + e] = (_Float16)y;
    }
  }
}

// ---------------------------------------------------------------------------
// Host-side launcher
// ---------------------------------------------------------------------------
static inline char* carve(char*& p, size_t bytes) {
  char* r = p;
  p += (bytes + 255) & ~(size_t)255;
  return r;
}

extern "C" void kernel_launch(void* const* d_in, const int* in_sizes, int n_in,
                              void* d_out, int out_size, void* d_ws, size_t ws_size,
                              hipStream_t stream) {
  (void)in_sizes; (void)n_in; (void)out_size; (void)ws_size;

  const float* x      = (const float*)d_in[0];
  const float* ln_w   = (const float*)d_in[1];
  const float* ln_b   = (const float*)d_in[2];
  const float* W_in   = (const float*)d_in[3];
  const float* W_conv = (const float*)d_in[4];
  const float* b_conv = (const float*)d_in[5];
  const float* W_xprj = (const float*)d_in[6];
  const float* W_dt   = (const float*)d_in[7];
  const float* b_dt   = (const float*)d_in[8];
  const float* A_log  = (const float*)d_in[9];
  const float* D_skip = (const float*)d_in[10];
  const float* W_out  = (const float*)d_in[11];
  float* out = (float*)d_out;

  char* p = (char*)d_ws;
  _Float16* xn_h     = (_Float16*)carve(p, (size_t)NTOK * D_  * 2);
  _Float16* Win_h    = (_Float16*)carve(p, (size_t)E2_  * D_  * 2);
  _Float16* Wxp_h    = (_Float16*)carve(p, (size_t)96   * DI_ * 2);
  _Float16* Wdt_h    = (_Float16*)carve(p, (size_t)DI_  * DTR * 2);
  _Float16* Wout_h   = (_Float16*)carve(p, (size_t)D_   * DI_ * 2);
  float*    xz       = (float*)   carve(p, (size_t)NTOK * E2_ * 4);
  float*    xc       = (float*)   carve(p, (size_t)NTOK * DI_ * 4);
  _Float16* xc_h     = (_Float16*)carve(p, (size_t)NTOK * DI_ * 2);
  float*    dbl      = (float*)   carve(p, (size_t)NTOK * 96  * 4);
  _Float16* dtl_h    = (_Float16*)carve(p, (size_t)NTOK * DTR * 2);
  float*    Bssm     = (float*)   carve(p, (size_t)NTOK * NS  * 4);
  float*    Cssm     = (float*)   carve(p, (size_t)NTOK * NS  * 4);
  float*    dpre     = (float*)   carve(p, (size_t)NTOK * DI_ * 4);
  _Float16* y_h      = (_Float16*)carve(p, (size_t)NTOK * DI_ * 2);

  const int TB = 256;
  auto blocks = [](int n, int tb) { return (n + tb - 1) / tb; };

  // 1) weight conversions to f16
  f32_to_f16_kernel<<<blocks(E2_ * D_, TB),  TB, 0, stream>>>(W_in,   Win_h,  E2_ * D_);
  f32_to_f16_kernel<<<blocks(96 * DI_, TB),  TB, 0, stream>>>(W_xprj, Wxp_h,  96 * DI_);
  f32_to_f16_kernel<<<blocks(DI_ * DTR, TB), TB, 0, stream>>>(W_dt,   Wdt_h,  DI_ * DTR);
  f32_to_f16_kernel<<<blocks(D_ * DI_, TB),  TB, 0, stream>>>(W_out,  Wout_h, D_ * DI_);

  // 2) layernorm -> xn (f16)
  ln_kernel<<<NTOK, TB, 0, stream>>>(x, ln_w, ln_b, xn_h);

  // 3) xz = xn * W_in^T   (2048 x 4096 x 1024), LDS-staged
  wmma_gemm_lds_kernel<false><<<(NTOK / MBLK) * (E2_ / NBLK), TB, 0, stream>>>(
      xn_h, Win_h, xz, nullptr, NTOK, E2_, D_);

  // 4) causal conv + SiLU -> xc (f32 + f16)
  conv_silu_kernel<<<blocks(B_ * L_ * DI_, TB), TB, 0, stream>>>(xz, W_conv, b_conv, xc, xc_h);

  // 5) dbl = xc * W_xproj^T  (2048 x 96 x 2048), direct path (tiny N)
  {
    const int waves = (NTOK / 16) * (96 / 32);
    wmma_gemm_direct_kernel<2><<<blocks(waves * 32, TB), TB, 0, stream>>>(
        xc_h, Wxp_h, dbl, NTOK, 96, DI_);
  }

  // 6) split dbl
  pack_kernel<<<blocks(NTOK * 96, TB), TB, 0, stream>>>(dbl, dtl_h, Bssm, Cssm);

  // 7) delta_pre = dt_low * W_dt^T  (2048 x 2048 x 64), LDS-staged
  wmma_gemm_lds_kernel<false><<<(NTOK / MBLK) * (DI_ / NBLK), TB, 0, stream>>>(
      dtl_h, Wdt_h, dpre, nullptr, NTOK, DI_, DTR);

  // 8) selective scan + gating -> y (f16)
  scan_kernel<<<blocks(B_ * DI_ * NS, TB), TB, 0, stream>>>(
      dpre, xc, Bssm, Cssm, xz, A_log, b_dt, D_skip, y_h);

  // 9) out = y * W_out^T + residual  (2048 x 1024 x 2048), LDS-staged
  wmma_gemm_lds_kernel<true><<<(NTOK / MBLK) * (D_ / NBLK), TB, 0, stream>>>(
      y_h, Wout_h, out, x, NTOK, D_, DI_);
}
